// RetroCrossAttention_40819369181901
// MI455X (gfx1250) — matrix operand order repaired and checked by reading, not compiled, not run on hardware
//
#include <hip/hip_runtime.h>
#include <hip/hip_bf16.h>

typedef __attribute__((ext_vector_type(16))) __bf16 v16bf;
typedef __attribute__((ext_vector_type(8)))  __bf16 v8bf;
typedef __attribute__((ext_vector_type(8)))  float  v8f;
typedef __attribute__((ext_vector_type(4)))  int    v4i;

#define B_  2
#define L_  32
#define K_  2
#define R_  64
#define M_  128
#define DM  2048
#define DQ  2048
#define H_  16
#define DH  128
#define BL  (B_*L_)
#define TOK 8192   /* B*L*K*R == B*L*M */

#if defined(__HIP_DEVICE_COMPILE__) && __has_builtin(__builtin_amdgcn_global_load_async_to_lds_b128)
#define ASYNC_LDS 1
#else
#define ASYNC_LDS 0
#endif

static __device__ inline void async_wait_all() {
#if ASYNC_LDS
#if __has_builtin(__builtin_amdgcn_s_wait_asynccnt)
  __builtin_amdgcn_s_wait_asynccnt(0);
#else
  asm volatile("s_wait_asynccnt 0x0" ::: "memory");
#endif
#endif
}

// copy 16 bf16 (32B) global -> LDS; async (2x b128) when available
static __device__ inline void cp16(__bf16* lds, const __bf16* g) {
#if ASYNC_LDS
  typedef __attribute__((address_space(1))) v4i GV4;
  typedef __attribute__((address_space(3))) v4i LV4;
  __builtin_amdgcn_global_load_async_to_lds_b128((GV4*)g, (LV4*)lds, 0, 0);
  __builtin_amdgcn_global_load_async_to_lds_b128((GV4*)(g + 8), (LV4*)(lds + 8), 0, 0);
#else
  *(v16bf*)lds = *(const v16bf*)g;
#endif
}

static __device__ inline v16bf cat16(v8bf lo, v8bf hi) {
  v16bf r;
#pragma unroll
  for (int i = 0; i < 8; ++i) { r[i] = lo[i]; r[i + 8] = hi[i]; }
  return r;
}
static __device__ inline v8f zero8() {
  v8f r;
#pragma unroll
  for (int i = 0; i < 8; ++i) r[i] = 0.0f;
  return r;
}

// ---------------- RMSNorm + cast to bf16 (one block per token) --------------
__global__ __launch_bounds__(256) void rmsnorm_kernel(
    const float* __restrict__ x, const float* __restrict__ ln_scale,
    __bf16* __restrict__ xn)
{
  __shared__ float red[256];
  const size_t tok = blockIdx.x;
  const float* xr = x + tok * DM;
  float ss = 0.f;
  for (int i = threadIdx.x; i < DM; i += 256) { float v = xr[i]; ss += v * v; }
  red[threadIdx.x] = ss;
  __syncthreads();
  for (int s = 128; s > 0; s >>= 1) {
    if ((int)threadIdx.x < s) red[threadIdx.x] += red[threadIdx.x + s];
    __syncthreads();
  }
  const float rms = sqrtf(red[0]) * 0.022097086912079608f; // *1/sqrt(2048)
  const float inv = 1.0f / (rms + 1e-8f);
  __bf16* xo = xn + tok * DM;
  for (int i = threadIdx.x; i < DM; i += 256)
    xo[i] = (__bf16)(xr[i] * inv * ln_scale[i]);
}

// ---------------- flat f32 -> bf16 cast -------------------------------------
__global__ __launch_bounds__(256) void cast_bf16_kernel(
    const float* __restrict__ in, __bf16* __restrict__ out, int n)
{
  int i = (blockIdx.x * 256 + threadIdx.x) * 4;
  if (i + 3 < n) {
    float4 v = *(const float4*)(in + i);
    out[i] = (__bf16)v.x; out[i+1] = (__bf16)v.y;
    out[i+2] = (__bf16)v.z; out[i+3] = (__bf16)v.w;
  } else {
    for (; i < n; ++i) out[i] = (__bf16)in[i];
  }
}

// ---------------- tiled bf16 GEMM: C = A(row-major) * W^T(row-major) --------
// 128x128 block tile, K-stage 64, double-buffered async LDS staging,
// 8 waves; each wave = 32x64 (2x4 WMMA tiles).
// MODE 0: q epilogue  (scatter to (bl,k,h,r,d) bf16)
// MODE 1: kv epilogue (k -> (bl,h,m,d) bf16 ; v -> transposed (bl,h,d,m) bf16)
// MODE 2: out epilogue (f32 + residual)
template <int MODE>
__global__ __launch_bounds__(256) void gemm_bf16_kernel(
    const __bf16* __restrict__ A, const __bf16* __restrict__ W, int Ncols,
    __bf16* __restrict__ out0, __bf16* __restrict__ out1,
    float* __restrict__ outf, const float* __restrict__ resid)
{
  __shared__ alignas(32) __bf16 As[2][128][64];
  __shared__ alignas(32) __bf16 Bs[2][128][64];
  const int m0 = blockIdx.y * 128;
  const int n0 = blockIdx.x * 128;
  const int tid = threadIdx.x;
  const int lane = tid & 31, w = tid >> 5;
  const int lm = lane & 15, hi = lane >> 4;
  const int wr = w & 3, wc = w >> 2;
  const int NK = DM / 64;

  // per-thread staging coordinates (2 chunks of 16 bf16 per tile)
  const int row0 = tid >> 2,         kc0 = (tid & 3) << 4;
  const int row1 = (tid + 256) >> 2, kc1 = ((tid + 256) & 3) << 4;

  v8f acc[2][4];
#pragma unroll
  for (int i = 0; i < 2; ++i)
#pragma unroll
    for (int j = 0; j < 4; ++j) acc[i][j] = zero8();

  auto issue_stage = [&](int kt) {
    const int k0 = kt * 64;
    const int buf = kt & 1;
    cp16(&As[buf][row0][kc0], A + (size_t)(m0 + row0) * DM + k0 + kc0);
    cp16(&Bs[buf][row0][kc0], W + (size_t)(n0 + row0) * DM + k0 + kc0);
    cp16(&As[buf][row1][kc1], A + (size_t)(m0 + row1) * DM + k0 + kc1);
    cp16(&Bs[buf][row1][kc1], W + (size_t)(n0 + row1) * DM + k0 + kc1);
    if (k0 + 128 < DM) {  // warm L2 two stages ahead
      __builtin_prefetch(A + (size_t)(m0 + row0) * DM + k0 + 128 + kc0, 0, 1);
      __builtin_prefetch(W + (size_t)(n0 + row0) * DM + k0 + 128 + kc0, 0, 1);
    }
  };

  issue_stage(0);
  for (int kt = 0; kt < NK; ++kt) {
    async_wait_all();
    __syncthreads();
    if (kt + 1 < NK) issue_stage(kt + 1);
    const int buf = kt & 1;
#pragma unroll
    for (int ks = 0; ks < 64; ks += 32) {
      v16bf a[2], b[4];
#pragma unroll
      for (int rt = 0; rt < 2; ++rt) {
        int r = wr * 32 + rt * 16 + lm;
        a[rt] = cat16(*(const v8bf*)&As[buf][r][ks + hi * 8],
                      *(const v8bf*)&As[buf][r][ks + 16 + hi * 8]);
      }
#pragma unroll
      for (int ct = 0; ct < 4; ++ct) {
        int n = wc * 64 + ct * 16 + lm;
        b[ct] = cat16(*(const v8bf*)&Bs[buf][n][ks + hi * 8],
                      *(const v8bf*)&Bs[buf][n][ks + 16 + hi * 8]);
      }
#pragma unroll
      for (int rt = 0; rt < 2; ++rt)
#pragma unroll
        for (int ct = 0; ct < 4; ++ct)
          acc[rt][ct] = __builtin_amdgcn_wmma_f32_16x16x32_bf16(
              false, a[rt], false, b[ct], (short)0, acc[rt][ct], false, false);
    }
  }

#pragma unroll
  for (int rt = 0; rt < 2; ++rt)
#pragma unroll
    for (int ct = 0; ct < 4; ++ct) {
      int colg = n0 + wc * 64 + ct * 16 + lm;
#pragma unroll
      for (int j = 0; j < 8; ++j) {
        int rowg = m0 + wr * 32 + rt * 16 + hi * 8 + j;
        float v = acc[rt][ct][j];
        if (MODE == 0) {
          int bl = rowg / (K_ * R_); int kr = rowg % (K_ * R_);
          int kk = kr / R_; int r = kr % R_;
          int h = colg / DH, d = colg % DH;
          out0[((((size_t)bl * K_ + kk) * H_ + h) * R_ + r) * DH + d] = (__bf16)v;
        } else if (MODE == 1) {
          int bl = rowg / M_; int m = rowg % M_;
          if (colg < DQ) {
            int h = colg / DH, d = colg % DH;
            out0[(((size_t)bl * H_ + h) * M_ + m) * DH + d] = (__bf16)v;
          } else {
            int c2 = colg - DQ; int h = c2 / DH, d = c2 % DH;
            out1[(((size_t)bl * H_ + h) * DH + d) * M_ + m] = (__bf16)v;  // V^T
          }
        } else {
          size_t idx = (size_t)rowg * DM + colg;
          outf[idx] = v + resid[idx];
        }
      }
    }
  (void)Ncols;
}

// ---------------- fused attention per (b,l,k,h): S -> softmax -> P*V --------
__global__ __launch_bounds__(256) void attention_kernel(
    const __bf16* __restrict__ q,     // [BL][K][H][R][DH]
    const __bf16* __restrict__ kmat,  // [BL][H][M][DH]
    const __bf16* __restrict__ vT,    // [BL][H][DH][M]
    const float* __restrict__ mask,   // [BL][M]
    const float* __restrict__ bias_table, // [32][H]
    __bf16* __restrict__ ctx,         // [TOK][DQ], col = h*DH+d
    float* __restrict__ attn)         // [BL][K][H][R][M]
{
  __shared__ float Sf[R_][M_];                 // 32 KB
  __shared__ alignas(32) __bf16 Pb[R_][M_];    // 16 KB
  const int bx = blockIdx.x;
  const int h = bx % H_;
  const int kk = (bx / H_) % K_;
  const int bl = bx / (H_ * K_);
  const int tid = threadIdx.x;
  const int lane = tid & 31, w = tid >> 5;
  const int lm = lane & 15, hi = lane >> 4;
  const int wr = w & 3, wc = w >> 2;

  const __bf16* qb = q + (((size_t)bl * K_ + kk) * H_ + h) * R_ * DH;
  const __bf16* kb = kmat + ((size_t)bl * H_ + h) * M_ * DH;
  const __bf16* vb = vT + ((size_t)bl * H_ + h) * DH * M_;

  // ---- S = q @ k^T  (64 x 128, K = DH) -------------------------------------
  v8f sacc[4];
#pragma unroll
  for (int ct = 0; ct < 4; ++ct) sacc[ct] = zero8();
#pragma unroll
  for (int kk0 = 0; kk0 < DH; kk0 += 32) {
    const __bf16* ap = qb + (size_t)(wr * 16 + lm) * DH + kk0;
    v16bf a = cat16(*(const v8bf*)(ap + hi * 8), *(const v8bf*)(ap + 16 + hi * 8));
#pragma unroll
    for (int ct = 0; ct < 4; ++ct) {
      const __bf16* bp = kb + (size_t)(wc * 64 + ct * 16 + lm) * DH + kk0;
      v16bf b = cat16(*(const v8bf*)(bp + hi * 8), *(const v8bf*)(bp + 16 + hi * 8));
      sacc[ct] = __builtin_amdgcn_wmma_f32_16x16x32_bf16(
          false, a, false, b, (short)0, sacc[ct], false, false);
    }
  }
#pragma unroll
  for (int ct = 0; ct < 4; ++ct) {
    int col = wc * 64 + ct * 16 + lm;
#pragma unroll
    for (int j = 0; j < 8; ++j) Sf[wr * 16 + hi * 8 + j][col] = sacc[ct][j];
  }
  __syncthreads();

  // ---- softmax with rel-pos bias + mask (one thread per row) ---------------
  if (tid < R_) {
    int r = tid;
    const float* mrow = mask + (size_t)bl * M_;
    float mx = -3.0e38f;
    for (int m = 0; m < M_; ++m) {
      int rel = m - r;                               // clip(rel,-15,16)+15
      int bkt = rel < -15 ? 0 : (rel > 16 ? 31 : rel + 15);
      float val = Sf[r][m] + bias_table[bkt * H_ + h] +
                  (1.0f - mrow[m]) * -10000.0f;
      Sf[r][m] = val;
      mx = fmaxf(mx, val);
    }
    float s = 0.f;
    for (int m = 0; m < M_; ++m) { float e = __expf(Sf[r][m] - mx); s += e; Sf[r][m] = e; }
    float inv = 1.0f / s;
    float* arow = attn + ((((size_t)bl * K_ + kk) * H_ + h) * R_ + r) * M_;
    for (int m = 0; m < M_; ++m) {
      float p = Sf[r][m] * inv;
      Pb[r][m] = (__bf16)p;
      arow[m] = p;
    }
  }
  __syncthreads();

  // ---- O = P @ V  (64 x 128, K = M; V stored transposed) -------------------
  v8f oacc[4];
#pragma unroll
  for (int ct = 0; ct < 4; ++ct) oacc[ct] = zero8();
#pragma unroll
  for (int kk0 = 0; kk0 < M_; kk0 += 32) {
    const __bf16* ap = &Pb[wr * 16 + lm][kk0];
    v16bf a = cat16(*(const v8bf*)(ap + hi * 8), *(const v8bf*)(ap + 16 + hi * 8));
#pragma unroll
    for (int ct = 0; ct < 4; ++ct) {
      const __bf16* bp = vb + (size_t)(wc * 64 + ct * 16 + lm) * M_ + kk0;
      v16bf b = cat16(*(const v8bf*)(bp + hi * 8), *(const v8bf*)(bp + 16 + hi * 8));
      oacc[ct] = __builtin_amdgcn_wmma_f32_16x16x32_bf16(
          false, a, false, b, (short)0, oacc[ct], false, false);
    }
  }
#pragma unroll
  for (int ct = 0; ct < 4; ++ct) {
    int d = wc * 64 + ct * 16 + lm;
#pragma unroll
    for (int j = 0; j < 8; ++j) {
      int r = wr * 16 + hi * 8 + j;
      size_t tokrow = ((size_t)bl * K_ + kk) * R_ + r;
      ctx[tokrow * DQ + (size_t)h * DH + d] = (__bf16)oacc[ct][j];
    }
  }
}

// ---------------- host-side launch ------------------------------------------
extern "C" void kernel_launch(void* const* d_in, const int* in_sizes, int n_in,
                              void* d_out, int out_size, void* d_ws, size_t ws_size,
                              hipStream_t stream) {
  const float* x     = (const float*)d_in[0];
  const float* hid   = (const float*)d_in[1];
  const float* mask  = (const float*)d_in[2];
  const float* lns   = (const float*)d_in[3];
  const float* q_w   = (const float*)d_in[4];
  const float* kv_w  = (const float*)d_in[5];
  const float* out_w = (const float*)d_in[6];
  const float* bias  = (const float*)d_in[7];

  char* ws = (char*)d_ws;
  size_t off = 0;
  auto alloc = [&](size_t bytes) -> void* {
    void* p = ws + off;
    off += (bytes + 255) & ~(size_t)255;
    return p;
  };
  __bf16* xn   = (__bf16*)alloc((size_t)TOK * DM * 2);
  __bf16* hb   = (__bf16*)alloc((size_t)TOK * DM * 2);
  __bf16* qwb  = (__bf16*)alloc((size_t)DQ * DM * 2);
  __bf16* kvwb = (__bf16*)alloc((size_t)2 * DQ * DM * 2);
  __bf16* owb  = (__bf16*)alloc((size_t)DM * DQ * 2);
  __bf16* qb   = (__bf16*)alloc((size_t)TOK * DQ * 2);
  __bf16* kbf  = (__bf16*)alloc((size_t)BL * H_ * M_ * DH * 2);
  __bf16* vtb  = (__bf16*)alloc((size_t)BL * H_ * DH * M_ * 2);
  __bf16* ctx  = (__bf16*)alloc((size_t)TOK * DQ * 2);

  float* outp  = (float*)d_out;
  float* attnp = outp + (size_t)TOK * DM;

  rmsnorm_kernel<<<TOK, 256, 0, stream>>>(x, lns, xn);
  cast_bf16_kernel<<<(TOK * DM) / 1024, 256, 0, stream>>>(hid, hb, TOK * DM);
  cast_bf16_kernel<<<(DQ * DM) / 1024, 256, 0, stream>>>(q_w, qwb, DQ * DM);
  cast_bf16_kernel<<<(2 * DQ * DM) / 1024, 256, 0, stream>>>(kv_w, kvwb, 2 * DQ * DM);
  cast_bf16_kernel<<<(DM * DQ) / 1024, 256, 0, stream>>>(out_w, owb, DM * DQ);

  dim3 gq(DQ / 128, TOK / 128);
  gemm_bf16_kernel<0><<<gq, 256, 0, stream>>>(xn, qwb, DQ, qb, nullptr, nullptr, nullptr);
  dim3 gkv(2 * DQ / 128, TOK / 128);
  gemm_bf16_kernel<1><<<gkv, 256, 0, stream>>>(hb, kvwb, 2 * DQ, kbf, vtb, nullptr, nullptr);

  attention_kernel<<<BL * K_ * H_, 256, 0, stream>>>(qb, kbf, vtb, mask, bias, ctx, attnp);

  dim3 go(DM / 128, TOK / 128);
  gemm_bf16_kernel<2><<<go, 256, 0, stream>>>(ctx, owb, DM, nullptr, nullptr, outp, x);
}
